// CVIM_91139206021556
// MI455X (gfx1250) — matrix operand order repaired and checked by reading, not compiled, not run on hardware
//
#include <hip/hip_runtime.h>
#include <hip/hip_bf16.h>

// Bidirectional cross-attention block for MI455X (gfx1250, wave32, WMMA + TDM).
//   1) transpose-convert 1x1-conv weights to f16 (c,d) layout
//   2) pointwise GEMM  h = x @ w1^T + b1           (WMMA, f32->f16 A on the fly)
//   3) depthwise k=3 conv -> Q/V in f16 (+ Q^T layout for score B-fragments)
//   4) S  = Q_l Q_r^T * C^-0.5 ; ST = Q_r Q_l^T * C^-0.5   (WMMA, TDM-staged B)
//   5) in-place row softmax of S and ST (LDS tree reductions)
//   6) F_r2l = P(S) @ V_r ; F_l2r = P(ST) @ V_l    (WMMA, TDM-staged B)
//   7) out = x + F @ w3^T + b3                     (WMMA, f32 out, residual)
// Workspace requirement: ~259 MB (see layout in kernel_launch).

#define BB 8
#define TT 2048
#define CC 512

typedef __attribute__((ext_vector_type(16))) _Float16 v16h;
typedef __attribute__((ext_vector_type(8)))  _Float16 v8h;
typedef __attribute__((ext_vector_type(8)))  float    v8f;
typedef __attribute__((ext_vector_type(4)))  unsigned int u32x4;
typedef __attribute__((ext_vector_type(8)))  int i32x8;
typedef __attribute__((ext_vector_type(4)))  int i32x4;

// LDS B-tile: 32 K-rows x 64 N-halves, padded by TDM to 144B pitch (72 halves)
// so ds_load_b128 stays 16B-aligned and bank-strides at 36 DWORDs.
#define BTILE_PITCH 72

// ---------------- WMMA fragment helpers (ISA 7.12.2 layouts) ----------------

// A-matrix 16x32 f16: lane%16 = row M; lane<16 holds K {0..7,16..23},
// lane>=16 holds K {8..15,24..31}. Two 16B loads per lane.
static __device__ __forceinline__ v16h load_a_f16(const _Float16* __restrict__ A, int lda) {
  const int lane = threadIdx.x & 31;
  const int row  = lane & 15;
  const int kb   = (lane >> 4) << 3;           // 0 or 8
  const _Float16* p = A + (size_t)row * lda + kb;
  v8h lo = *(const v8h*)(p);                    // K = kb .. kb+7
  v8h hi = *(const v8h*)(p + 16);               // K = kb+16 .. kb+23
  v16h r;
#pragma unroll
  for (int i = 0; i < 8; ++i) { r[i] = lo[i]; r[8 + i] = hi[i]; }
  return r;
}

// Same A fragment but sourced from f32 memory, converted in-register.
static __device__ __forceinline__ v16h load_a_f32(const float* __restrict__ A, int lda) {
  const int lane = threadIdx.x & 31;
  const int row  = lane & 15;
  const int kb   = (lane >> 4) << 3;
  const float* p = A + (size_t)row * lda + kb;
  v16h r;
#pragma unroll
  for (int i = 0; i < 8; ++i) { r[i] = (_Float16)p[i]; r[8 + i] = (_Float16)p[i + 16]; }
  return r;
}

// B-matrix 32x16 f16 from global: lane%16 = col N; lane<16 holds K {0..7,16..23},
// lane>=16 holds K {8..15,24..31} (matches A).
static __device__ __forceinline__ v16h load_b_f16(const _Float16* __restrict__ Bm, int ldb) {
  const int lane = threadIdx.x & 31;
  const int n  = lane & 15;
  const int kb = (lane >> 4) << 3;
  const _Float16* p = Bm + (size_t)kb * ldb + n;
  v16h r;
#pragma unroll
  for (int i = 0; i < 8; ++i) { r[i] = p[(size_t)i * ldb]; r[8 + i] = p[(size_t)(16 + i) * ldb]; }
  return r;
}

// B-matrix 32x16 f16 from the padded LDS tile (col in halves), same K order as A.
static __device__ __forceinline__ v16h load_b_lds(const _Float16* bt, int col) {
  const int lane = threadIdx.x & 31;
  const int kb = (lane >> 4) << 3;
  const _Float16* p = bt + (size_t)kb * BTILE_PITCH + col + (lane & 15);
  v16h r;
#pragma unroll
  for (int i = 0; i < 8; ++i) { r[i] = p[i * BTILE_PITCH]; r[8 + i] = p[(16 + i) * BTILE_PITCH]; }
  return r;
}

static __device__ __forceinline__ v8f wmma16(v16h a, v16h b, v8f c) {
  return __builtin_amdgcn_wmma_f32_16x16x32_f16(false, a, false, b, (short)0, c, false, false);
}

// C/D 16x16 f32: VGPR r holds M=r (lanes 0-15) / M=8+r (lanes 16-31), N=lane&15.
static __device__ __forceinline__ void store_tile_f16(_Float16* __restrict__ D, int ldd,
                                                      const v8f& a, float mul) {
  const int lane = threadIdx.x & 31;
  const int n = lane & 15, mo = (lane >> 4) << 3;
#pragma unroll
  for (int r = 0; r < 8; ++r) D[(size_t)(mo + r) * ldd + n] = (_Float16)(a[r] * mul);
}

// ---------------- Tensor Data Mover (ISA ch.8 D# layout) ----------------
// 2D f16 tile load: tile_x contiguous halves, tile_y rows with row stride
// `stride0` (halves). LDS destination gets 4 pad DWORDs after every 32 DWORDs
// (128B rows -> 144B pitch). Issued by one wave; tracked with TENSORcnt.
static __device__ __forceinline__ void tdm_load_2d(unsigned lds_off, const void* gptr,
                                                   unsigned tile_x, unsigned tile_y,
                                                   unsigned dim0, unsigned dim1,
                                                   unsigned long long stride0) {
  const unsigned long long ga = (unsigned long long)(size_t)gptr;
  u32x4 g0;
  g0[0] = 1u;                                             // count=1, user mode
  g0[1] = lds_off;                                        // lds_addr (bytes)
  g0[2] = (unsigned)(ga & 0xffffffffu);                   // global_addr[31:0]
  g0[3] = (unsigned)((ga >> 32) & 0x1ffffffu) | (2u << 30); // [56:32] | type=2
  i32x8 g1;
  g1[0] = (int)((1u << 16) |                              // data_size = 2 bytes
                (1u << 20) |                              // pad_enable
                (4u << 22) |                              // pad_interval: 32 DWORDs
                (3u << 25));                              // pad_amount: 4 DWORDs
  g1[1] = (int)((dim0 & 0xffffu) << 16);                  // tensor_dim0[15:0]
  g1[2] = (int)(((dim0 >> 16) & 0xffffu) | ((dim1 & 0xffffu) << 16));
  g1[3] = (int)(((dim1 >> 16) & 0xffffu) | ((tile_x & 0xffffu) << 16));
  g1[4] = (int)(tile_y & 0xffffu);                        // tile_dim1 (tile_dim2=0)
  g1[5] = (int)(unsigned)(stride0 & 0xffffffffu);         // dim0_stride[31:0]
  g1[6] = (int)(unsigned)((stride0 >> 32) & 0xffffu);     // dim0_stride[47:32]
  g1[7] = 0;
  const i32x4 z4 = {0, 0, 0, 0};
#if defined(__clang_major__) && __clang_major__ >= 23
  const i32x8 z8 = {0, 0, 0, 0, 0, 0, 0, 0};
  __builtin_amdgcn_tensor_load_to_lds(g0, g1, z4, z4, z8, 0);
#else
  __builtin_amdgcn_tensor_load_to_lds(g0, g1, z4, z4, 0);
#endif
}

static __device__ __forceinline__ unsigned lds_offset_of(const void* p) {
  // Generic pointer into LDS aperture: low 32 bits are the LDS byte offset.
  return (unsigned)(size_t)p;
}

// ---------------- kernels ----------------

// Transpose+convert a (C,C) f32 weight (d,c) into f16 (c,d) so it can serve
// as a row-major B-matrix (K=c, N=d).
__global__ __launch_bounds__(256) void k_cvtT(const float* __restrict__ W,
                                              _Float16* __restrict__ WT) {
  const int idx = blockIdx.x * 256 + threadIdx.x;   // idx = c*C + d
  const int c = idx / CC, d = idx % CC;
  WT[idx] = (_Float16)W[(size_t)d * CC + c];
}

// Pointwise 1x1 conv: H(B*T,C) f32 = X(B*T,C) f32 @ WT(C,C) f16 + b.
// Block tile 128x64, 8 waves in 4x2, each wave 32x32 (4 WMMA tiles).
// Weights are 512KB f16 -> resident in L2/WGP$, so direct global B loads.
__global__ __launch_bounds__(256) void k_pointwise(const float* __restrict__ X,
                                                   const _Float16* __restrict__ WT,
                                                   const float* __restrict__ bias,
                                                   float* __restrict__ H) {
  const int wave = threadIdx.x >> 5;
  const int wm = wave & 3, wn = wave >> 2;
  const int m0 = blockIdx.x * 128 + wm * 32;
  const int n0 = blockIdx.y * 64 + wn * 32;
  v8f acc[2][2] = {};
  for (int kk = 0; kk < CC; kk += 32) {
    if (kk + 64 < CC) {
      __builtin_prefetch((const void*)(X + (size_t)m0 * CC + kk + 64), 0, 1);
      __builtin_prefetch((const void*)(WT + (size_t)(kk + 64) * CC + n0), 0, 1);
    }
    v16h a0 = load_a_f32(X + (size_t)m0 * CC + kk, CC);
    v16h a1 = load_a_f32(X + (size_t)(m0 + 16) * CC + kk, CC);
    v16h b0 = load_b_f16(WT + (size_t)kk * CC + n0, CC);
    v16h b1 = load_b_f16(WT + (size_t)kk * CC + n0 + 16, CC);
    acc[0][0] = wmma16(a0, b0, acc[0][0]);
    acc[0][1] = wmma16(a0, b1, acc[0][1]);
    acc[1][0] = wmma16(a1, b0, acc[1][0]);
    acc[1][1] = wmma16(a1, b1, acc[1][1]);
  }
  const int lane = threadIdx.x & 31;
  const int n = lane & 15, mo = (lane >> 4) << 3;
#pragma unroll
  for (int i = 0; i < 2; ++i)
#pragma unroll
    for (int j = 0; j < 2; ++j) {
      const float bj = bias[n0 + j * 16 + n];
#pragma unroll
      for (int r = 0; r < 8; ++r)
        H[(size_t)(m0 + i * 16 + mo + r) * CC + n0 + j * 16 + n] = acc[i][j][r] + bj;
    }
}

// Depthwise conv k=3 pad=1 over T on H, writes f16 Y (B,T,C) and optionally
// the transposed f16 YT (B,C,T) used as score B-matrix.
__global__ __launch_bounds__(256) void k_dw(const float* __restrict__ H,
                                            const float* __restrict__ W2,
                                            const float* __restrict__ B2,
                                            _Float16* __restrict__ Y,
                                            _Float16* __restrict__ YT) {
  const size_t idx = (size_t)blockIdx.x * 256 + threadIdx.x;  // over B*T*C
  const int d = (int)(idx % CC);
  const int t = (int)((idx / CC) % TT);
  const size_t b = idx / ((size_t)TT * CC);
  float acc = B2[d] + H[idx] * W2[d * 3 + 1];
  if (t > 0)      acc += H[idx - CC] * W2[d * 3 + 0];
  if (t < TT - 1) acc += H[idx + CC] * W2[d * 3 + 2];
  const _Float16 h = (_Float16)acc;
  Y[idx] = h;
  if (YT) YT[(b * CC + d) * TT + t] = h;
}

// Generic batched f16 GEMM: D(M,N) f16 = A(M,K) f16 @ B(K,N) f16 * mul.
// B tiles (32x64 f16) are staged in LDS by the Tensor Data Mover, double
// buffered: wave 0 issues TDM for tile i+1, waits TENSORcnt<=1 (tile i done),
// a workgroup barrier publishes it; trailing barrier protects buffer reuse.
__global__ __launch_bounds__(256) void k_gemm_f16(const _Float16* __restrict__ A,
                                                  const _Float16* __restrict__ B,
                                                  _Float16* __restrict__ D,
                                                  int K, int lda, int ldb, int ldd,
                                                  long long strideA, long long strideB,
                                                  long long strideD, float mul) {
  __shared__ __align__(16) _Float16 bt[2][32 * BTILE_PITCH];
  const size_t b = blockIdx.z;
  A += b * strideA; B += b * strideB; D += b * strideD;
  const int wave = threadIdx.x >> 5;
  const int wm = wave & 3, wn = wave >> 2;
  const int m0 = blockIdx.x * 128 + wm * 32;
  const int nblk = blockIdx.y * 64;
  const int n0 = nblk + wn * 32;

  if (wave == 0)
    tdm_load_2d(lds_offset_of(&bt[0][0]), B + nblk, 64, 32,
                (unsigned)ldb, (unsigned)K, (unsigned long long)ldb);

  v8f acc[2][2] = {};
  int cur = 0;
  for (int kk = 0; kk < K; kk += 32) {
    if (wave == 0) {
      if (kk + 32 < K) {
        tdm_load_2d(lds_offset_of(&bt[cur ^ 1][0]), B + (size_t)(kk + 32) * ldb + nblk,
                    64, 32, (unsigned)ldb, (unsigned)K, (unsigned long long)ldb);
        __builtin_amdgcn_s_wait_tensorcnt(1);  // current tile has landed
      } else {
        __builtin_amdgcn_s_wait_tensorcnt(0);
      }
    }
    __syncthreads();
    if (kk + 64 < K)
      __builtin_prefetch((const void*)(A + (size_t)m0 * lda + kk + 64), 0, 1);
    v16h a0 = load_a_f16(A + (size_t)m0 * lda + kk, lda);
    v16h a1 = load_a_f16(A + (size_t)(m0 + 16) * lda + kk, lda);
    v16h b0 = load_b_lds(&bt[cur][0], wn * 32);
    v16h b1 = load_b_lds(&bt[cur][0], wn * 32 + 16);
    acc[0][0] = wmma16(a0, b0, acc[0][0]);
    acc[0][1] = wmma16(a0, b1, acc[0][1]);
    acc[1][0] = wmma16(a1, b0, acc[1][0]);
    acc[1][1] = wmma16(a1, b1, acc[1][1]);
    __syncthreads();  // all reads of bt[cur] done before TDM overwrites it
    cur ^= 1;
  }
#pragma unroll
  for (int i = 0; i < 2; ++i)
#pragma unroll
    for (int j = 0; j < 2; ++j)
      store_tile_f16(D + (size_t)(m0 + i * 16) * ldd + n0 + j * 16, ldd, acc[i][j], mul);
}

// In-place row softmax over T=2048 f16 values; one block per row.
__global__ __launch_bounds__(256) void k_softmax(_Float16* __restrict__ S) {
  __shared__ float red[256];
  _Float16* p = S + (size_t)blockIdx.x * TT;
  const int t0 = threadIdx.x;
  float x[8];
#pragma unroll
  for (int i = 0; i < 8; ++i) x[i] = (float)p[t0 + 256 * i];
  float m = -1e30f;
#pragma unroll
  for (int i = 0; i < 8; ++i) m = fmaxf(m, x[i]);
  red[t0] = m; __syncthreads();
  for (int s = 128; s > 0; s >>= 1) {
    if (t0 < s) red[t0] = fmaxf(red[t0], red[t0 + s]);
    __syncthreads();
  }
  m = red[0]; __syncthreads();
  float l = 0.f;
#pragma unroll
  for (int i = 0; i < 8; ++i) { x[i] = __expf(x[i] - m); l += x[i]; }
  red[t0] = l; __syncthreads();
  for (int s = 128; s > 0; s >>= 1) {
    if (t0 < s) red[t0] += red[t0 + s];
    __syncthreads();
  }
  const float rinv = 1.0f / red[0];
#pragma unroll
  for (int i = 0; i < 8; ++i) p[t0 + 256 * i] = (_Float16)(x[i] * rinv);
}

// Output projection: Out(B*T,C) f32 = X + F(B*T,C) f16 @ WT(C,C) f16 + b.
__global__ __launch_bounds__(256) void k_out(const _Float16* __restrict__ F,
                                             const _Float16* __restrict__ WT,
                                             const float* __restrict__ bias,
                                             const float* __restrict__ X,
                                             float* __restrict__ Out) {
  const int wave = threadIdx.x >> 5;
  const int wm = wave & 3, wn = wave >> 2;
  const int m0 = blockIdx.x * 128 + wm * 32;
  const int n0 = blockIdx.y * 64 + wn * 32;
  v8f acc[2][2] = {};
  for (int kk = 0; kk < CC; kk += 32) {
    if (kk + 64 < CC) {
      __builtin_prefetch((const void*)(F + (size_t)m0 * CC + kk + 64), 0, 1);
      __builtin_prefetch((const void*)(WT + (size_t)(kk + 64) * CC + n0), 0, 1);
    }
    v16h a0 = load_a_f16(F + (size_t)m0 * CC + kk, CC);
    v16h a1 = load_a_f16(F + (size_t)(m0 + 16) * CC + kk, CC);
    v16h b0 = load_b_f16(WT + (size_t)kk * CC + n0, CC);
    v16h b1 = load_b_f16(WT + (size_t)kk * CC + n0 + 16, CC);
    acc[0][0] = wmma16(a0, b0, acc[0][0]);
    acc[0][1] = wmma16(a0, b1, acc[0][1]);
    acc[1][0] = wmma16(a1, b0, acc[1][0]);
    acc[1][1] = wmma16(a1, b1, acc[1][1]);
  }
  const int lane = threadIdx.x & 31;
  const int n = lane & 15, mo = (lane >> 4) << 3;
#pragma unroll
  for (int i = 0; i < 2; ++i)
#pragma unroll
    for (int j = 0; j < 2; ++j) {
      const float bj = bias[n0 + j * 16 + n];
#pragma unroll
      for (int r = 0; r < 8; ++r) {
        const size_t off = (size_t)(m0 + i * 16 + mo + r) * CC + n0 + j * 16 + n;
        Out[off] = X[off] + acc[i][j][r] + bj;
      }
    }
}

// ---------------- host-side orchestration ----------------

extern "C" void kernel_launch(void* const* d_in, const int* in_sizes, int n_in,
                              void* d_out, int out_size, void* d_ws, size_t ws_size,
                              hipStream_t stream) {
  const float* x_l = (const float*)d_in[0];
  const float* x_r = (const float*)d_in[1];
  const float* lp1_w1 = (const float*)d_in[2];  const float* lp1_b1 = (const float*)d_in[3];
  const float* lp1_w2 = (const float*)d_in[4];  const float* lp1_b2 = (const float*)d_in[5];
  const float* rp1_w1 = (const float*)d_in[6];  const float* rp1_b1 = (const float*)d_in[7];
  const float* rp1_w2 = (const float*)d_in[8];  const float* rp1_b2 = (const float*)d_in[9];
  const float* lp2_w1 = (const float*)d_in[10]; const float* lp2_b1 = (const float*)d_in[11];
  const float* lp2_w2 = (const float*)d_in[12]; const float* lp2_b2 = (const float*)d_in[13];
  const float* rp2_w1 = (const float*)d_in[14]; const float* rp2_b1 = (const float*)d_in[15];
  const float* rp2_w2 = (const float*)d_in[16]; const float* rp2_b2 = (const float*)d_in[17];
  const float* lp3_w  = (const float*)d_in[18]; const float* lp3_b  = (const float*)d_in[19];
  const float* rp3_w  = (const float*)d_in[20]; const float* rp3_b  = (const float*)d_in[21];

  float* out_l = (float*)d_out;
  float* out_r = out_l + (size_t)BB * TT * CC;

  // Workspace carve-up (~259 MB).
  char* ws = (char*)d_ws;
  const size_t szW   = (size_t)CC * CC * sizeof(_Float16);        // 512 KB
  const size_t szBTC = (size_t)BB * TT * CC * sizeof(_Float16);   // 16 MB
  const size_t szBTT = (size_t)BB * TT * TT * sizeof(_Float16);   // 64 MB
  _Float16* wT1l = (_Float16*)ws;            ws += szW;
  _Float16* wT1r = (_Float16*)ws;            ws += szW;
  _Float16* wT2l = (_Float16*)ws;            ws += szW;
  _Float16* wT2r = (_Float16*)ws;            ws += szW;
  _Float16* wT3l = (_Float16*)ws;            ws += szW;
  _Float16* wT3r = (_Float16*)ws;            ws += szW;
  float*    h    = (float*)ws;               ws += (size_t)BB * TT * CC * sizeof(float);
  _Float16* Ql   = (_Float16*)ws;            ws += szBTC;
  _Float16* Qr   = (_Float16*)ws;            ws += szBTC;
  _Float16* Vl   = (_Float16*)ws;            ws += szBTC;
  _Float16* Vr   = (_Float16*)ws;            ws += szBTC;
  _Float16* QlT  = (_Float16*)ws;            ws += szBTC;
  _Float16* QrT  = (_Float16*)ws;            ws += szBTC;
  _Float16* S    = (_Float16*)ws;            ws += szBTT;
  _Float16* ST   = (_Float16*)ws;            ws += szBTT;
  // Q^T buffers are dead after the score GEMMs; reuse them for F outputs.
  _Float16* Fr2l = QlT;
  _Float16* Fl2r = QrT;

  const dim3 blk(256);
  const float scale = 0.04419417382415922f;  // C^-0.5

  // 1) weight transposes to f16 (c,d)
  const int gW = (CC * CC) / 256;
  k_cvtT<<<gW, blk, 0, stream>>>(lp1_w1, wT1l);
  k_cvtT<<<gW, blk, 0, stream>>>(rp1_w1, wT1r);
  k_cvtT<<<gW, blk, 0, stream>>>(lp2_w1, wT2l);
  k_cvtT<<<gW, blk, 0, stream>>>(rp2_w1, wT2r);
  k_cvtT<<<gW, blk, 0, stream>>>(lp3_w,  wT3l);
  k_cvtT<<<gW, blk, 0, stream>>>(rp3_w,  wT3r);

  // 2,3) four projection chains (pointwise GEMM + depthwise); h is reused.
  const dim3 gP((BB * TT) / 128, CC / 64, 1);
  const int gD = (int)(((size_t)BB * TT * CC) / 256);
  k_pointwise<<<gP, blk, 0, stream>>>(x_l, wT1l, lp1_b1, h);
  k_dw<<<gD, blk, 0, stream>>>(h, lp1_w2, lp1_b2, Ql, QlT);
  k_pointwise<<<gP, blk, 0, stream>>>(x_r, wT1r, rp1_b1, h);
  k_dw<<<gD, blk, 0, stream>>>(h, rp1_w2, rp1_b2, Qr, QrT);
  k_pointwise<<<gP, blk, 0, stream>>>(x_l, wT2l, lp2_b1, h);
  k_dw<<<gD, blk, 0, stream>>>(h, lp2_w2, lp2_b2, Vl, ((_Float16*)0));
  k_pointwise<<<gP, blk, 0, stream>>>(x_r, wT2r, rp2_b1, h);
  k_dw<<<gD, blk, 0, stream>>>(h, rp2_w2, rp2_b2, Vr, ((_Float16*)0));

  // 4) scores in both orientations (attn and attn^T are the same GEMM swapped)
  const dim3 gS(TT / 128, TT / 64, BB);
  k_gemm_f16<<<gS, blk, 0, stream>>>(Ql, QrT, S, CC, CC, TT, TT,
                                     (long long)TT * CC, (long long)CC * TT,
                                     (long long)TT * TT, scale);
  k_gemm_f16<<<gS, blk, 0, stream>>>(Qr, QlT, ST, CC, CC, TT, TT,
                                     (long long)TT * CC, (long long)CC * TT,
                                     (long long)TT * TT, scale);

  // 5) row softmax (r->l over s on S; l->r over t on ST), in place
  k_softmax<<<BB * TT, blk, 0, stream>>>(S);
  k_softmax<<<BB * TT, blk, 0, stream>>>(ST);

  // 6) P @ V
  const dim3 gA(TT / 128, CC / 64, BB);
  k_gemm_f16<<<gA, blk, 0, stream>>>(S, Vr, Fr2l, TT, TT, CC, CC,
                                     (long long)TT * TT, (long long)TT * CC,
                                     (long long)TT * CC, 1.0f);
  k_gemm_f16<<<gA, blk, 0, stream>>>(ST, Vl, Fl2r, TT, TT, CC, CC,
                                     (long long)TT * TT, (long long)TT * CC,
                                     (long long)TT * CC, 1.0f);

  // 7) output projections with residual
  const dim3 gO((BB * TT) / 128, CC / 64, 1);
  k_out<<<gO, blk, 0, stream>>>(Fr2l, wT3l, lp3_b, x_l, out_l);
  k_out<<<gO, blk, 0, stream>>>(Fl2r, wT3r, rp3_b, x_r, out_r);
}
